// Network_20950850470063
// MI455X (gfx1250) — matrix operand order repaired and checked by reading, not compiled, not run on hardware
//
#include <hip/hip_runtime.h>

// ---------------------------------------------------------------------------
// SLAYER 2-layer SNN for MI455X (gfx1250, wave32, WMMA).
//   gemm1 (f16 WMMA, K=32768, 4-buffer pipeline) -> FIR(100) -> spike IIR ->
//   gemm2 (f16 WMMA, K=416) -> FIR(100) -> spike IIR -> d_out
// ---------------------------------------------------------------------------

typedef __attribute__((ext_vector_type(16))) _Float16 v16h;
typedef __attribute__((ext_vector_type(8)))  _Float16 v8h;
typedef __attribute__((ext_vector_type(8)))  float    v8f;

constexpr int kNB    = 8;            // batch
constexpr int kCin   = 2*128*128;    // 32768
constexpr int kT     = 300;
constexpr int kTpad  = 304;          // 19 tiles of 16
constexpr int kTT    = 19;           // t tiles per batch
constexpr int kO1    = 410;
constexpr int kO1pad = 448;          // 7 blocks of 64
constexpr int kO2    = 10;
constexpr int kK2    = 416;          // K for gemm2, 13 * 32
constexpr int kKlen  = 100;          // FIR taps
constexpr int kNtTiles = kNB * kTT;  // 152

// A-fragment (16x32 f16): lane L holds row M=L%16; elements 0..7 = K at
// (L/16)*8, elements 8..15 = same + 16.  Two 16-byte loads.
__device__ __forceinline__ v16h load_afrag(const _Float16* p) {
  union { v16h v; v8h h[2]; } u;
  u.h[0] = *(const v8h*)(p);
  u.h[1] = *(const v8h*)(p + 16);
  return u.v;
}

// One k-step's worth of fragments for a 64(M) x 16(NT) wave tile.
struct Frags {
  v16h b;
  v16h a[4];
};

__device__ __forceinline__ Frags load_frags(const _Float16* __restrict__ b0,
                                            const _Float16* __restrict__ a0,
                                            int k) {
  Frags f;
  f.b = load_afrag(b0 + k);
#pragma unroll
  for (int i = 0; i < 4; ++i)
    f.a[i] = load_afrag(a0 + (size_t)(16*i)*kCin + k);
  return f;
}

// ---------------------------------------------------------------------------
// x (n, i, t) f32  ->  xT (n, tpad, i) f16   (LDS tile transpose, zero-pad t)
// grid (kCin/32, ceil(kTpad/32), kNB), block (32, 8)
// ---------------------------------------------------------------------------
__global__ void convert_x_kernel(const float* __restrict__ x,
                                 _Float16* __restrict__ xT) {
  __shared__ float tile[32][33];
  const int n  = blockIdx.z;
  const int i0 = blockIdx.x * 32;
  const int t0 = blockIdx.y * 32;
#pragma unroll
  for (int j = 0; j < 4; ++j) {
    const int li = threadIdx.y + 8*j;
    const int lt = threadIdx.x;
    const int gt = t0 + lt;
    float v = 0.0f;
    if (gt < kT) v = x[((size_t)n*kCin + (i0 + li))*kT + gt];  // coalesced in t
    tile[li][lt] = v;
  }
  __syncthreads();
#pragma unroll
  for (int j = 0; j < 4; ++j) {
    const int lt = threadIdx.y + 8*j;
    const int li = threadIdx.x;
    const int gt = t0 + lt;
    if (gt < kTpad)
      xT[((size_t)n*kTpad + gt)*kCin + (i0 + li)] = (_Float16)tile[li][lt];
  }
}

// w1 (410, 32768) f32 -> w1f (448, 32768) f16, zero rows >= 410
__global__ void convert_w1_kernel(const float* __restrict__ w1,
                                  _Float16* __restrict__ w1f) {
  const size_t idx = (size_t)blockIdx.x * blockDim.x + threadIdx.x;
  if (idx >= (size_t)kO1pad * kCin) return;
  const int row = (int)(idx / kCin);
  const int col = (int)(idx % kCin);
  const float v = (row < kO1) ? w1[(size_t)row*kCin + col] : 0.0f;
  w1f[idx] = (_Float16)v;
}

// w2 (10, 410) f32 -> w2f (16, 416) f16, zero-padded
__global__ void convert_w2_kernel(const float* __restrict__ w2,
                                  _Float16* __restrict__ w2f) {
  const int idx = blockIdx.x * blockDim.x + threadIdx.x;
  if (idx >= 16 * kK2) return;
  const int row = idx / kK2;
  const int col = idx % kK2;
  const float v = (row < kO2 && col < kO1) ? w2[row*kO1 + col] : 0.0f;
  w2f[idx] = (_Float16)v;
}

// ---------------------------------------------------------------------------
// GEMM1: u1[n,o,t] = sum_i w1[o,i] * x[n,i,t]
// wave = 64(M) x 16(NT) tile, K loop 32768/32.
// Four fixed fragment buffers (step 128): each buffer is refilled right after
// its WMMAs issue and not consumed again for 3 WMMA groups, so every load has
// ~12 WMMAs of slack and no wait has to drain to zero.  grid (7, 152), block 32.
// ---------------------------------------------------------------------------
__global__ __launch_bounds__(32)
void gemm1_kernel(const _Float16* __restrict__ w1f,
                  const _Float16* __restrict__ xT,
                  float* __restrict__ u1) {
  const int lane = threadIdx.x;
  const int col  = lane & 15;
  const int half = lane >> 4;
  const int ob   = blockIdx.x * 64;
  const int nt   = blockIdx.y;
  const int n    = nt / kTT;
  const int tt   = (nt % kTT) * 16;

  const _Float16* b0 = xT  + (size_t)(n*kTpad + tt + col) * kCin + half*8;
  const _Float16* a0 = w1f + (size_t)(ob + col)            * kCin + half*8;

  v8f acc[4] = {v8f{}, v8f{}, v8f{}, v8f{}};

  auto mm = [&](const Frags& f) {
#pragma unroll
    for (int a = 0; a < 4; ++a) {
      acc[a] = __builtin_amdgcn_wmma_f32_16x16x32_f16(
          false, f.a[a], false, f.b, (short)0, acc[a], false, false);
    }
  };

  // Tail prefetches read <= ~512B past the operand arrays; that lands in the
  // next workspace carve and the values are never consumed.
  Frags f0 = load_frags(b0, a0, 0);
  Frags f1 = load_frags(b0, a0, 32);
  Frags f2 = load_frags(b0, a0, 64);
  Frags f3 = load_frags(b0, a0, 96);
  for (int k = 0; k < kCin; k += 128) {      // 256 iterations, 16 WMMAs each
    mm(f0);  f0 = load_frags(b0, a0, k + 128);
    mm(f1);  f1 = load_frags(b0, a0, k + 160);
    mm(f2);  f2 = load_frags(b0, a0, k + 192);
    mm(f3);  f3 = load_frags(b0, a0, k + 224);
  }

  const int t = tt + col;
  if (t < kT) {
#pragma unroll
    for (int a = 0; a < 4; ++a) {
#pragma unroll
      for (int r = 0; r < 8; ++r) {
        const int o = ob + a*16 + r + 8*half;   // C layout: row = r + 8*(lane>=16)
        if (o < kO1) u1[((size_t)n*kO1 + o)*kT + t] = acc[a][r];
      }
    }
  }
}

// ---------------------------------------------------------------------------
// GEMM2: u2[n,o,t] = sum_k w2f[o,k] * s1f[n,t,k]   (K = 416, M tile = 16)
// grid 152, block 32.
// ---------------------------------------------------------------------------
__global__ __launch_bounds__(32)
void gemm2_kernel(const _Float16* __restrict__ w2f,
                  const _Float16* __restrict__ s1f,
                  float* __restrict__ u2) {
  const int lane = threadIdx.x;
  const int col  = lane & 15;
  const int half = lane >> 4;
  const int nt   = blockIdx.x;
  const int n    = nt / kTT;
  const int tt   = (nt % kTT) * 16;

  const _Float16* b0 = s1f + (size_t)(n*kTpad + tt + col) * kK2 + half*8;
  const _Float16* a0 = w2f + (size_t)col * kK2 + half*8;

  v8f acc = {};
#pragma unroll
  for (int k = 0; k < kK2; k += 32) {
    const v16h b  = load_afrag(b0 + k);
    const v16h av = load_afrag(a0 + k);
    acc = __builtin_amdgcn_wmma_f32_16x16x32_f16(
        false, av, false, b, (short)0, acc, false, false);
  }

  const int t = tt + col;
  if (t < kT) {
#pragma unroll
    for (int r = 0; r < 8; ++r) {
      const int o = r + 8*half;
      if (o < kO2) u2[((size_t)n*kO2 + o)*kT + t] = acc[r];
    }
  }
}

// ---------------------------------------------------------------------------
// Causal FIR with SRM alpha kernel eps(m) = (m/10) * exp(1 - m/10), 100 taps.
// One block per (n,o) row; row + kernel cached in LDS.
// ---------------------------------------------------------------------------
__global__ void fir_kernel(const float* __restrict__ z, float* __restrict__ out) {
  __shared__ float row[kT];
  __shared__ float kern[kKlen];
  const int r = blockIdx.x;
  const float* zp = z + (size_t)r * kT;
  for (int t = threadIdx.x; t < kT; t += blockDim.x) row[t] = zp[t];
  for (int m = threadIdx.x; m < kKlen; m += blockDim.x)
    kern[m] = ((float)m * 0.1f) * __expf(1.0f - (float)m * 0.1f);
  __syncthreads();
  for (int t = threadIdx.x; t < kT; t += blockDim.x) {
    float acc = 0.0f;
    const int mmax = t < (kKlen - 1) ? t : (kKlen - 1);
    for (int m = 0; m <= mmax; ++m) acc += kern[m] * row[t - m];
    out[(size_t)r*kT + t] = acc;
  }
}

// ---------------------------------------------------------------------------
// Sequential refractory + threshold per neuron:
//   y[n] = A*(y[n-1] + K*p[n-1]);  s = (u + y >= theta);  p = A*p + s
// A = exp(-1), K = -2*theta*e*0.1 = -2e, theta = 10.
// ---------------------------------------------------------------------------
constexpr float kAref = 0.36787944117144233f;   // exp(-1)
constexpr float kKref = -5.43656365691809f;     // -2*e

__global__ void spike1_kernel(const float* __restrict__ u,
                              _Float16* __restrict__ s1f) {
  const int idx = blockIdx.x * blockDim.x + threadIdx.x;
  if (idx >= kNB * kO1) return;
  const int n = idx / kO1;
  const int o = idx % kO1;
  const float* up = u + (size_t)idx * kT;
  float p = 0.0f, y = 0.0f;
  for (int t = 0; t < kT; ++t) {
    y = kAref * (y + kKref * p);
    const float s = (up[t] + y >= 10.0f) ? 1.0f : 0.0f;
    p = kAref * p + s;
    s1f[((size_t)n*kTpad + t)*kK2 + o] = (_Float16)s;  // B-layout for gemm2
  }
}

__global__ void spike2_kernel(const float* __restrict__ u,
                              float* __restrict__ out) {
  const int idx = blockIdx.x * blockDim.x + threadIdx.x;
  if (idx >= kNB * kO2) return;
  const float* up = u + (size_t)idx * kT;
  float p = 0.0f, y = 0.0f;
  for (int t = 0; t < kT; ++t) {
    y = kAref * (y + kKref * p);
    const float s = (up[t] + y >= 10.0f) ? 1.0f : 0.0f;
    p = kAref * p + s;
    out[(size_t)idx*kT + t] = s;                       // (n, o, t) f32
  }
}

// ---------------------------------------------------------------------------

extern "C" void kernel_launch(void* const* d_in, const int* in_sizes, int n_in,
                              void* d_out, int out_size, void* d_ws, size_t ws_size,
                              hipStream_t stream) {
  const float* x  = (const float*)d_in[0];   // (8, 2,128,128, 300) f32
  const float* w1 = (const float*)d_in[1];   // (410, 32768) f32
  const float* w2 = (const float*)d_in[2];   // (10, 410) f32
  float* out = (float*)d_out;                // (8, 10, 300) f32

  // workspace carve-out (256B aligned): ~199 MB total
  char* ws = (char*)d_ws;
  auto carve = [&](size_t bytes) -> char* {
    char* p = ws;
    ws += (bytes + 255) & ~(size_t)255;
    return p;
  };
  _Float16* xT  = (_Float16*)carve((size_t)kNB*kTpad*kCin * 2);   // 159.4 MB
  _Float16* w1f = (_Float16*)carve((size_t)kO1pad*kCin    * 2);   //  29.4 MB
  _Float16* s1f = (_Float16*)carve((size_t)kNB*kTpad*kK2  * 2);   //   2.0 MB
  _Float16* w2f = (_Float16*)carve((size_t)16*kK2         * 2);
  float*    u1  = (float*)   carve((size_t)kNB*kO1*kT * 4);
  float*    p1  = (float*)   carve((size_t)kNB*kO1*kT * 4);
  float*    u2  = (float*)   carve((size_t)kNB*kO2*kT * 4);
  float*    p2  = (float*)   carve((size_t)kNB*kO2*kT * 4);

  // zero s1f so the (t >= 300) / (o >= 410) pads are clean for gemm2
  hipMemsetAsync(s1f, 0, (size_t)kNB*kTpad*kK2 * 2, stream);

  convert_x_kernel<<<dim3(kCin/32, (kTpad + 31)/32, kNB), dim3(32, 8), 0, stream>>>(x, xT);
  {
    const size_t total = (size_t)kO1pad * kCin;
    convert_w1_kernel<<<(unsigned)((total + 255)/256), 256, 0, stream>>>(w1, w1f);
  }
  convert_w2_kernel<<<(16*kK2 + 255)/256, 256, 0, stream>>>(w2, w2f);

  gemm1_kernel<<<dim3(kO1pad/64, kNtTiles), 32, 0, stream>>>(w1f, xT, u1);
  fir_kernel<<<kNB*kO1, 128, 0, stream>>>(u1, p1);
  spike1_kernel<<<(kNB*kO1 + 255)/256, 256, 0, stream>>>(p1, s1f);

  gemm2_kernel<<<kNtTiles, 32, 0, stream>>>(w2f, s1f, u2);
  fir_kernel<<<kNB*kO2, 128, 0, stream>>>(u2, p2);
  spike2_kernel<<<1, 128, 0, stream>>>(p2, out);
}
